// MessageBuildingLayerLSH_70720931496069
// MI455X (gfx1250) — compile-verified, matrix operand order checked by reading, not run on hardware
//
#include <hip/hip_runtime.h>
#include <hip/hip_bf16.h>

typedef __attribute__((ext_vector_type(16))) _Float16 v16h;
typedef __attribute__((ext_vector_type(8)))  float    v8f;

#define B_DIM   16
#define N_PTS   12800
#define DMSG    128
#define DNODE   512
#define NBINS   100
#define NHALF   50
#define NKEY    199      // keys can reach argmax(99) + 99 for masked points
#define BINSZ   128

// ---------------------------------------------------------------------------
// 1) LSH projection + argmax over [mul, -mul]  -> bin key per point
//    Coalesced via per-wave 32x32 LDS transpose tiles; codebook in LDS.
// ---------------------------------------------------------------------------
__global__ __launch_bounds__(256) void proj_argmax_kernel(
    const float* __restrict__ x_msg, const float* __restrict__ codebook,
    const unsigned char* __restrict__ msk, int* __restrict__ bin_idx) {
  __shared__ float cb[DMSG * NHALF];      // 6400 floats = 25.6 KB
  __shared__ float tiles[8][32 * 33];     // 8 waves * 32x32 (+pad) = 33.8 KB
  const int t = threadIdx.x;
  const int wave = t >> 5, lane = t & 31;
  const int b = blockIdx.y;
  const int n0 = blockIdx.x * 256;

  for (int i = t; i < DMSG * NHALF; i += 256) {
    int d = i / NHALF, j = i - d * NHALF;
    cb[i] = codebook[d * NBINS + j];      // first 50 columns of (128,100)
  }
  __syncthreads();

  const size_t rowbase = (size_t)b * N_PTS + n0 + wave * 32;
  float acc[NHALF];
#pragma unroll
  for (int j = 0; j < NHALF; ++j) acc[j] = 0.0f;

  for (int d0 = 0; d0 < DMSG; d0 += 32) {
    // coalesced: lane l loads column (d0+l) of 32 consecutive rows
    for (int r = 0; r < 32; ++r)
      tiles[wave][r * 33 + lane] = x_msg[(rowbase + r) * DMSG + d0 + lane];
    __syncthreads();
    for (int dd = 0; dd < 32; ++dd) {
      float xv = tiles[wave][lane * 33 + dd];   // conflict-free (stride 33)
      const float* c = cb + (d0 + dd) * NHALF;  // uniform -> LDS broadcast
#pragma unroll
      for (int j = 0; j < NHALF; ++j) acc[j] += xv * c[j];
    }
    __syncthreads();
  }

  float best = acc[0]; int bi = 0;
#pragma unroll
  for (int j = 1; j < NHALF; ++j) if (acc[j] > best) { best = acc[j]; bi = j; }
#pragma unroll
  for (int j = 0; j < NHALF; ++j) if (-acc[j] > best) { best = -acc[j]; bi = NHALF + j; }

  const size_t n = (size_t)b * N_PTS + n0 + wave * 32 + lane;
  bin_idx[n] = bi + (msk[n] ? 0 : (NBINS - 1));
}

// ---------------------------------------------------------------------------
// 2) per-batch histogram over 199 keys + exclusive scan -> offsets
// ---------------------------------------------------------------------------
__global__ __launch_bounds__(256) void hist_offsets_kernel(
    const int* __restrict__ bin_idx, int* __restrict__ offsets) {
  __shared__ int cnt[NKEY];
  const int b = blockIdx.x, t = threadIdx.x;
  for (int i = t; i < NKEY; i += 256) cnt[i] = 0;
  __syncthreads();
  for (int n = t; n < N_PTS; n += 256) atomicAdd(&cnt[bin_idx[(size_t)b * N_PTS + n]], 1);
  __syncthreads();
  if (t == 0) {
    int s = 0;
    for (int k = 0; k < NKEY; ++k) { offsets[b * NKEY + k] = s; s += cnt[k]; }
  }
}

// ---------------------------------------------------------------------------
// 3) stable counting-sort scatter: one wave32 per (batch,key); in-order scan
//    with ballot prefix => deterministic stable permutation. Also emits the
//    bins_split (as float) and msk_f_binned outputs.
// ---------------------------------------------------------------------------
__global__ __launch_bounds__(32) void stable_scatter_kernel(
    const int* __restrict__ bin_idx, const int* __restrict__ offsets,
    const unsigned char* __restrict__ msk, int* __restrict__ flat,
    float* __restrict__ out_bins, float* __restrict__ out_msk) {
  const int blk = blockIdx.x;
  const int b = blk / NKEY, key = blk - b * NKEY;
  const int lane = threadIdx.x;
  int cnt = offsets[b * NKEY + key];
  const unsigned lt = (1u << lane) - 1u;
  for (int n0 = 0; n0 < N_PTS; n0 += 32) {
    const int n = n0 + lane;
    const bool flag = (bin_idx[(size_t)b * N_PTS + n] == key);
    const unsigned m32 = (unsigned)__ballot(flag);   // wave32: low 32 bits
    if (flag) {
      const int pos = cnt + __popc(m32 & lt);
      const size_t o = (size_t)b * N_PTS + pos;
      flat[o] = n;
      out_bins[o] = (float)n;
      out_msk[o] = msk[(size_t)b * N_PTS + n] ? 1.0f : 0.0f;
    }
    cnt += __popc(m32);
  }
}

// ---------------------------------------------------------------------------
// 4) gather x_node rows (512 floats) by permutation; float4 (B128) copies
// ---------------------------------------------------------------------------
__global__ __launch_bounds__(128) void gather_node_kernel(
    const float* __restrict__ x_node, const int* __restrict__ flat,
    float* __restrict__ out_feat) {
  const int bk = blockIdx.x;                 // b*N + k
  const int b = bk / N_PTS;
  const int src = flat[bk];
  const float4* s = (const float4*)(x_node + ((size_t)b * N_PTS + src) * DNODE);
  float4* d = (float4*)(out_feat + (size_t)bk * DNODE);
  d[threadIdx.x] = s[threadIdx.x];
}

// ---------------------------------------------------------------------------
// 5) per-bin Gaussian kernel matrix via WMMA f16->f32 Gram
//    One block per (b,bin): stage masked 128x128 tile as f16 in LDS,
//    8 waves each own a 16-row strip, 8 column tiles, K=128 in 4 WMMA steps.
// ---------------------------------------------------------------------------
__global__ __launch_bounds__(256) void dm_wmma_kernel(
    const float* __restrict__ x_msg, const unsigned char* __restrict__ msk,
    const int* __restrict__ flat, float* __restrict__ out_dm) {
  __shared__ _Float16 Xh[128 * 136];   // 34.8 KB, stride 136 halves
  __shared__ float norms[128];
  __shared__ float mv[128];
  const int bb = blockIdx.x;           // b*100 + bin
  const int b = bb / NBINS;
  const int t = threadIdx.x;
  const int* fl = flat + (size_t)b * N_PTS + (size_t)(bb - b * NBINS) * BINSZ;

  // stage masked tile (coalesced float4 reads of gathered rows)
  for (int it = 0; it < 16; ++it) {
    const int idx = t + it * 256;      // 4096 float4 slots
    const int row = idx >> 5, c4 = idx & 31;
    const int g = fl[row];
    const float4 v = *(const float4*)(x_msg + ((size_t)b * N_PTS + g) * DMSG + c4 * 4);
    const float m = msk[(size_t)b * N_PTS + g] ? 1.0f : 0.0f;
    _Float16* dst = Xh + row * 136 + c4 * 4;
    dst[0] = (_Float16)(v.x * m); dst[1] = (_Float16)(v.y * m);
    dst[2] = (_Float16)(v.z * m); dst[3] = (_Float16)(v.w * m);
    if (c4 == 0) mv[row] = m;
  }
  __syncthreads();
  if (t < 128) {                       // f32 row norms of the (f16) tile
    const _Float16* r = Xh + t * 136;
    float s = 0.0f;
    for (int d = 0; d < DMSG; ++d) { const float x = (float)r[d]; s += x * x; }
    norms[t] = s;
  }
  __syncthreads();

  const int wave = t >> 5, lane = t & 31;
  const int i0 = wave * 16;
  const int mrow = lane & 15;
  const int koff = (lane >> 4) << 3;   // lanes 16..31 hold K+8 per ISA layout
  v8f acc[8] = {};

  for (int k0 = 0; k0 < DMSG; k0 += 32) {
    v16h a;
    const _Float16* ra = Xh + (i0 + mrow) * 136 + k0 + koff;
#pragma unroll
    for (int v = 0; v < 8; ++v) {      // VGPR v: K = (v>=4)*16 + koff + (v&3)*2
      const int kk = ((v >> 2) << 4) + ((v & 3) << 1);
      a[2 * v] = ra[kk]; a[2 * v + 1] = ra[kk + 1];
    }
#pragma unroll
    for (int j = 0; j < 8; ++j) {
      v16h bf;
      const _Float16* rb = Xh + (j * 16 + mrow) * 136 + k0 + koff;
#pragma unroll
      for (int v = 0; v < 8; ++v) {
        const int kk = ((v >> 2) << 4) + ((v & 3) << 1);
        bf[2 * v] = rb[kk]; bf[2 * v + 1] = rb[kk + 1];
      }
      acc[j] = __builtin_amdgcn_wmma_f32_16x16x32_f16(
          false, a, false, bf, (short)0, acc[j], false, false);
    }
  }

  // epilogue: D2 = |xi|^2 + |xj|^2 - 2*G ; clip -> sqrt -> exp -> clip -> mask
  const size_t base = (size_t)bb * (BINSZ * BINSZ);
  const int mbase = i0 + ((lane < 16) ? 0 : 8);   // C/D layout: VGPR r -> M=r(+8)
  const int ncol = lane & 15;
#pragma unroll
  for (int j = 0; j < 8; ++j) {
    const int n = j * 16 + ncol;
    const float nb = norms[n], mn = mv[n];
#pragma unroll
    for (int r = 0; r < 8; ++r) {
      const int m = mbase + r;
      float d2 = norms[m] + nb - 2.0f * acc[j][r];
      d2 = fminf(fmaxf(d2, 1e-6f), 1e6f);
      float val = expf(-0.1f * sqrtf(d2));
      val = fminf(fmaxf(val, 0.0f), 1.0f);
      out_dm[base + (size_t)m * BINSZ + n] = val * mv[m] * mn;
    }
  }
}

// ---------------------------------------------------------------------------
extern "C" void kernel_launch(void* const* d_in, const int* in_sizes, int n_in,
                              void* d_out, int out_size, void* d_ws, size_t ws_size,
                              hipStream_t stream) {
  (void)in_sizes; (void)n_in; (void)out_size; (void)ws_size;
  const float* x_msg = (const float*)d_in[0];
  const float* x_node = (const float*)d_in[1];
  const unsigned char* msk = (const unsigned char*)d_in[2];   // bool array
  const float* codebook = (const float*)d_in[3];

  // outputs concatenated flat: bins_split, x_features_binned, dm, msk_f_binned
  float* out = (float*)d_out;
  float* out_bins = out;                                                // 204800
  float* out_feat = out + (size_t)B_DIM * N_PTS;                        // 104857600
  float* out_dm   = out_feat + (size_t)B_DIM * N_PTS * DNODE;           // 26214400
  float* out_msk  = out_dm + (size_t)B_DIM * NBINS * BINSZ * BINSZ;     // 204800

  char* w = (char*)d_ws;
  int* bin_idx = (int*)w;                                               // B*N ints
  int* offsets = (int*)(w + (size_t)B_DIM * N_PTS * 4);                 // B*NKEY ints
  int* flat    = (int*)(w + (size_t)B_DIM * N_PTS * 4 + (size_t)B_DIM * NKEY * 4);

  proj_argmax_kernel<<<dim3(N_PTS / 256, B_DIM), 256, 0, stream>>>(x_msg, codebook, msk, bin_idx);
  hist_offsets_kernel<<<B_DIM, 256, 0, stream>>>(bin_idx, offsets);
  stable_scatter_kernel<<<B_DIM * NKEY, 32, 0, stream>>>(bin_idx, offsets, msk, flat,
                                                         out_bins, out_msk);
  gather_node_kernel<<<B_DIM * N_PTS, 128, 0, stream>>>(x_node, flat, out_feat);
  dm_wmma_kernel<<<B_DIM * NBINS, 256, 0, stream>>>(x_msg, msk, flat, out_dm);
}
